// UltimateFastMQA_61383672594954
// MI455X (gfx1250) — matrix operand order repaired and checked by reading, not compiled, not run on hardware
//
#include <hip/hip_runtime.h>
#include <hip/hip_bf16.h>
#include <math.h>

// ---------------------------------------------------------------------------
// Problem constants (match reference)
// ---------------------------------------------------------------------------
#define HIDDEN   2048
#define HEADS    16
#define HD       128           // HEAD_DIM
#define MLA      32
#define SEQ      2048
#define BATCH    4
#define WIN      512
#define STEPW    256
#define NWIN     8             // SEQ / STEPW
#define MTOT     (BATCH*SEQ)   // 8192 rows
#define ATT_SCALE 0.08838834764831845f  // 1/sqrt(128)

typedef __bf16 bf16;
typedef __attribute__((ext_vector_type(16))) __bf16 v16bf;
typedef __attribute__((ext_vector_type(8)))  float  v8f;
typedef __attribute__((ext_vector_type(4)))  unsigned int u32x4;
typedef __attribute__((ext_vector_type(8)))  int i32x8;
typedef __attribute__((ext_vector_type(4)))  int i32x4;

// ---------------------------------------------------------------------------
// Tensor Data Mover: async stage of a 2D bf16 tile (rows x tile_d0 elems,
// row-major with row stride stride0 elems) into LDS at lds_off.
// D# groups per CDNA5 ISA. Builtin arity differs between
// ROCm 7.2 (5-arg) and upstream clang-23 (6-arg) — switch on __clang_major__.
// ---------------------------------------------------------------------------
__device__ __forceinline__ void tdm_load_tile_2d(unsigned lds_off,
                                                 const void* gptr,
                                                 unsigned tile_d0,
                                                 unsigned rows,
                                                 unsigned stride0,
                                                 unsigned tensor_d0)
{
    unsigned long long ga = (unsigned long long)gptr;
    u32x4 g0;
    g0[0] = 1u;                                     // count=1, user descriptor
    g0[1] = lds_off;                                // lds_addr
    g0[2] = (unsigned)(ga & 0xFFFFFFFFu);           // global_addr[31:0]
    g0[3] = (unsigned)((ga >> 32) & 0x01FFFFFFu)    // global_addr[56:32]
          | (2u << 30);                             // type=2 ("image")
    i32x8 g1;
    g1[0] = (int)(1u << 16);                        // wg_mask=0, data_size=1 (2B)
    g1[1] = (int)((tensor_d0 & 0xFFFFu) << 16);     // tensor_dim0[15:0] @63:48
    g1[2] = (int)(((tensor_d0 >> 16) & 0xFFFFu)     // tensor_dim0[31:16]
          | ((16384u & 0xFFFFu) << 16));            // tensor_dim1[15:0]
    g1[3] = (int)(((16384u >> 16) & 0xFFFFu)        // tensor_dim1[31:16]
          | ((tile_d0 & 0xFFFFu) << 16));           // tile_dim0
    g1[4] = (int)(rows & 0xFFFFu);                  // tile_dim1 = rows, dim2=0
    g1[5] = (int)stride0;                           // tensor_dim0_stride lo
    g1[6] = 0;                                      // stride hi, dim1_stride lo
    g1[7] = 0;
    i32x4 z4 = {0, 0, 0, 0};
#if defined(__clang_major__) && __clang_major__ >= 23
    i32x8 z8 = {0, 0, 0, 0, 0, 0, 0, 0};
    __builtin_amdgcn_tensor_load_to_lds(g0, g1, z4, z4, z8, 0);
#else
    __builtin_amdgcn_tensor_load_to_lds(g0, g1, z4, z4, 0);
#endif
}

// order-preserving float <-> uint key for atomic min/max
__device__ __forceinline__ unsigned fenc(float f) {
    unsigned u = __float_as_uint(f);
    return (u & 0x80000000u) ? ~u : (u | 0x80000000u);
}
__device__ __forceinline__ float fdec(unsigned k) {
    unsigned u = (k & 0x80000000u) ? (k & 0x7FFFFFFFu) : ~k;
    return __uint_as_float(u);
}
__device__ __forceinline__ unsigned umin2(unsigned a, unsigned b){return a<b?a:b;}
__device__ __forceinline__ unsigned umax2(unsigned a, unsigned b){return a>b?a:b;}

// ---------------------------------------------------------------------------
// Elementwise helpers
// ---------------------------------------------------------------------------
__global__ void cvt_f32_bf16_kernel(const float* __restrict__ in,
                                    bf16* __restrict__ out, size_t n)
{
    size_t i = (size_t)blockIdx.x * blockDim.x + threadIdx.x;
    size_t stride = (size_t)gridDim.x * blockDim.x;
    for (; i < n; i += stride) out[i] = (bf16)in[i];
}

__global__ void fill_f32_kernel(float* __restrict__ p, float v, size_t n)
{
    size_t i = (size_t)blockIdx.x * blockDim.x + threadIdx.x;
    size_t stride = (size_t)gridDim.x * blockDim.x;
    for (; i < n; i += stride) p[i] = v;
}

__global__ void init_mm_kernel(unsigned* mm)
{
    if (threadIdx.x == 0) {
        mm[0] = 0xFFFFFFFFu; mm[1] = 0u;   // K min/max keys
        mm[2] = 0xFFFFFFFFu; mm[3] = 0u;   // V min/max keys
    }
}

// ---------------------------------------------------------------------------
// GEMM: C[M,N] (f32) = A[M,K] (bf16) @ W[N,K]^T (bf16)   (weights row = out ch)
// Block tile 128x128, 256 threads = 8 waves, each wave 32x64 (2x4 WMMA tiles).
// K-step 64, TDM double-buffered LDS staging (waves 0/1 own the DMA queues),
// next tile's DMA overlaps current tile's 16 WMMAs via s_wait_tensorcnt(1).
// ---------------------------------------------------------------------------
__global__ __launch_bounds__(256)
void gemm_bf16_wt_kernel(const bf16* __restrict__ A, const bf16* __restrict__ W,
                         float* __restrict__ C, int M, int N, int K)
{
    const int bm = blockIdx.y * 128;
    const int bn = blockIdx.x * 128;
    const int tid  = threadIdx.x;
    const int wave = tid >> 5, lane = tid & 31;
    const int lh = lane & 15, hi = (lane >> 4) & 1;
    const int wm = (wave & 3) * 32;      // wave M offset within block
    const int wn = (wave >> 2) * 64;     // wave N offset within block

    __shared__ __align__(128) bf16 As[2][128][64];
    __shared__ __align__(128) bf16 Bs[2][128][64];

    v8f acc[2][4];
#pragma unroll
    for (int mt = 0; mt < 2; ++mt)
#pragma unroll
        for (int nt = 0; nt < 4; ++nt) acc[mt][nt] = (v8f){0,0,0,0,0,0,0,0};

    const bf16* Ab = A + (size_t)bm * K;
    const bf16* Wb = W + (size_t)bn * K;

    // prologue: stage K-step 0 into buffer 0
    if (wave == 0)
        tdm_load_tile_2d((unsigned)(size_t)&As[0][0][0], Ab, 64, 128, K, K);
    else if (wave == 1)
        tdm_load_tile_2d((unsigned)(size_t)&Bs[0][0][0], Wb, 64, 128, K, K);

    const int nsteps = K >> 6;           // K / 64
    for (int s = 0; s < nsteps; ++s) {
        const int cur = s & 1;
        const int nxt = cur ^ 1;
        if (s + 1 < nsteps) {
            if (wave == 0)
                tdm_load_tile_2d((unsigned)(size_t)&As[nxt][0][0],
                                 Ab + (size_t)(s + 1) * 64, 64, 128, K, K);
            else if (wave == 1)
                tdm_load_tile_2d((unsigned)(size_t)&Bs[nxt][0][0],
                                 Wb + (size_t)(s + 1) * 64, 64, 128, K, K);
            if (wave < 2) __builtin_amdgcn_s_wait_tensorcnt(1);
        } else {
            if (wave < 2) __builtin_amdgcn_s_wait_tensorcnt(0);
        }
        __syncthreads();                 // current buffer ready for all waves

#pragma unroll
        for (int ks = 0; ks < 2; ++ks) { // two 32-deep WMMA sub-steps
            v16bf af[2], bw[4];
#pragma unroll
            for (int mt = 0; mt < 2; ++mt)
                af[mt] = *(const v16bf*)&As[cur][wm + mt * 16 + lh][ks * 32 + hi * 16];
#pragma unroll
            for (int nt = 0; nt < 4; ++nt)
                bw[nt] = *(const v16bf*)&Bs[cur][wn + nt * 16 + lh][ks * 32 + hi * 16];
#pragma unroll
            for (int mt = 0; mt < 2; ++mt)
#pragma unroll
                for (int nt = 0; nt < 4; ++nt)
                    acc[mt][nt] = __builtin_amdgcn_wmma_f32_16x16x32_bf16(
                        false, af[mt], false, bw[nt], (short)0, acc[mt][nt],
                        false, false);
        }
        __syncthreads();                 // done reading before DMA overwrites
    }

#pragma unroll
    for (int mt = 0; mt < 2; ++mt)
#pragma unroll
        for (int nt = 0; nt < 4; ++nt)
#pragma unroll
            for (int r = 0; r < 8; ++r) {
                int row = bm + wm + mt * 16 + r + hi * 8;
                int col = bn + wn + nt * 16 + lh;
                C[(size_t)row * N + col] = acc[mt][nt][r];
            }
}

// ---------------------------------------------------------------------------
// RoPE
// ---------------------------------------------------------------------------
__global__ void rope_q_kernel(const float* __restrict__ Q,
                              bf16* __restrict__ out, size_t n)
{
    size_t i = (size_t)blockIdx.x * blockDim.x + threadIdx.x;
    size_t stride = (size_t)gridDim.x * blockDim.x;
    for (; i < n; i += stride) {
        int col = (int)(i % HIDDEN);
        int s   = (int)((i / HIDDEN) % SEQ);
        int d   = col & (HD - 1);
        int j   = d & 63;
        float inv = __expf(-9.210340371976184f * ((float)j * (1.0f / 64.0f)));
        float sn, cs;
        __sincosf((float)s * inv, &sn, &cs);
        float x   = Q[i];
        float rot = (d < 64) ? -Q[i + 64] : Q[i - 64];
        out[i] = (bf16)(x * cs + rot * sn);
    }
}

__global__ void rope_k_kernel(const float* __restrict__ Kin,
                              float* __restrict__ out, size_t n)
{
    size_t i = (size_t)blockIdx.x * blockDim.x + threadIdx.x;
    size_t stride = (size_t)gridDim.x * blockDim.x;
    for (; i < n; i += stride) {
        int d = (int)(i % HD);
        int s = (int)((i / HD) % SEQ);
        int j = d & 63;
        float inv = __expf(-9.210340371976184f * ((float)j * (1.0f / 64.0f)));
        float sn, cs;
        __sincosf((float)s * inv, &sn, &cs);
        float x   = Kin[i];
        float rot = (d < 64) ? -Kin[i + 64] : Kin[i - 64];
        out[i] = x * cs + rot * sn;
    }
}

// ---------------------------------------------------------------------------
// MLA: per token, 128 -> 32 (cw[32,128]) -> 128 (dw[128,32]); track min/max.
// ---------------------------------------------------------------------------
__global__ __launch_bounds__(128)
void compress_kernel(const float* __restrict__ in, const float* __restrict__ cw,
                     const float* __restrict__ dw, float* __restrict__ out,
                     unsigned* __restrict__ mm)
{
    const int token = blockIdx.x, tid = threadIdx.x;
    __shared__ float xv[HD];
    __shared__ float cc[MLA];
    __shared__ unsigned smn[4], smx[4];

    xv[tid] = in[(size_t)token * HD + tid];
    __syncthreads();
    if (tid < MLA) {
        float s = 0.f;
#pragma unroll 8
        for (int d = 0; d < HD; ++d) s += cw[tid * HD + d] * xv[d];
        cc[tid] = s;
    }
    __syncthreads();
    float o = 0.f;
#pragma unroll
    for (int m = 0; m < MLA; ++m) o += dw[tid * MLA + m] * cc[m];
    out[(size_t)token * HD + tid] = o;

    unsigned k = fenc(o), kmin = k, kmax = k;
#pragma unroll
    for (int off = 16; off; off >>= 1) {
        kmin = umin2(kmin, (unsigned)__shfl_xor((int)kmin, off));
        kmax = umax2(kmax, (unsigned)__shfl_xor((int)kmax, off));
    }
    if ((tid & 31) == 0) { smn[tid >> 5] = kmin; smx[tid >> 5] = kmax; }
    __syncthreads();
    if (tid == 0) {
        unsigned a = smn[0], b = smx[0];
        for (int i = 1; i < 4; ++i) { a = umin2(a, smn[i]); b = umax2(b, smx[i]); }
        atomicMin(&mm[0], a);
        atomicMax(&mm[1], b);
    }
}

__global__ void quant_kernel(const float* __restrict__ in, bf16* __restrict__ out,
                             const unsigned* __restrict__ mm, size_t n)
{
    float mn = fdec(mm[0]), mx = fdec(mm[1]);
    float sc = (mx - mn) * (1.0f / 255.0f);
    float invs = (sc != 0.f) ? 1.0f / sc : 0.f;
    size_t i = (size_t)blockIdx.x * blockDim.x + threadIdx.x;
    size_t stride = (size_t)gridDim.x * blockDim.x;
    for (; i < n; i += stride) {
        float q = rintf((in[i] - mn) * invs);
        q = fminf(fmaxf(q, 0.f), 255.f);
        out[i] = (bf16)(q * sc + mn);
    }
}

// ---------------------------------------------------------------------------
// Windowed sigmoid attention + cross-fade. One block = 128 query rows of one
// (batch, head, window). Streams 64-key chunks with double-buffered TDM K/V
// staging; WMMA Q@K^T -> sigmoid -> P (via LDS) -> WMMA P@V; sigmoid rowsums
// accumulate in LDS (ds_add_f32). Final: ctx += weight * acc / rowsum via
// 2-way global atomicAdd blend (deterministic: exactly two contributions).
// ---------------------------------------------------------------------------
__global__ __launch_bounds__(256)
void attn_kernel(const bf16* __restrict__ Qb, const bf16* __restrict__ Kq,
                 const bf16* __restrict__ Vq, float* __restrict__ ctx)
{
    const int idx = blockIdx.x;
    const int c = idx & 3;             // 128-row query chunk within window
    const int w = (idx >> 2) & 7;      // window
    const int h = (idx >> 5) & 15;     // head
    const int b = idx >> 9;            // batch
    const int win_start = w * STEPW;
    const int win_len = (SEQ - win_start < WIN) ? (SEQ - win_start) : WIN;
    if (c * 128 >= win_len) return;

    const int tid = threadIdx.x, wave = tid >> 5, lane = tid & 31;
    const int lh = lane & 15, hi = (lane >> 4) & 1;

    __shared__ __align__(128) bf16 Ks[2][64][HD];
    __shared__ __align__(128) bf16 Vs[2][64][HD];
    __shared__ __align__(128) bf16 Ps[8][16][32];
    __shared__ float rs[8][16];

    if (tid < 128) ((float*)rs)[tid] = 1e-8f;

    const int qs = win_start + c * 128 + wave * 16;
    const bf16* Kbase = Kq + ((size_t)b * SEQ + win_start) * HD;
    const bf16* Vbase = Vq + ((size_t)b * SEQ + win_start) * HD;

    // preload Q fragments (16 rows x 128 dims, bf16) for this wave
    v16bf qf[4];
    {
        const bf16* qp = Qb + ((size_t)b * SEQ + (qs + lh)) * HIDDEN
                            + h * HD + hi * 16;
        qf[0] = *(const v16bf*)(qp);
        qf[1] = *(const v16bf*)(qp + 32);
        qf[2] = *(const v16bf*)(qp + 64);
        qf[3] = *(const v16bf*)(qp + 96);
    }

    v8f acc[8];
#pragma unroll
    for (int nt = 0; nt < 8; ++nt) acc[nt] = (v8f){0,0,0,0,0,0,0,0};

    // prologue: stage chunk 0 into buffer 0
    if (wave == 0)
        tdm_load_tile_2d((unsigned)(size_t)&Ks[0][0][0], Kbase, HD, 64, HD, HD);
    else if (wave == 1)
        tdm_load_tile_2d((unsigned)(size_t)&Vs[0][0][0], Vbase, HD, 64, HD, HD);

    const int nchunks = win_len >> 6;    // 64-key chunks (4 or 8)
    for (int kc = 0; kc < nchunks; ++kc) {
        const int cur = kc & 1;
        const int nxt = cur ^ 1;
        if (kc + 1 < nchunks) {
            if (wave == 0)
                tdm_load_tile_2d((unsigned)(size_t)&Ks[nxt][0][0],
                                 Kbase + (size_t)(kc + 1) * 64 * HD, HD, 64, HD, HD);
            else if (wave == 1)
                tdm_load_tile_2d((unsigned)(size_t)&Vs[nxt][0][0],
                                 Vbase + (size_t)(kc + 1) * 64 * HD, HD, 64, HD, HD);
            if (wave < 2) __builtin_amdgcn_s_wait_tensorcnt(1);
        } else {
            if (wave < 2) __builtin_amdgcn_s_wait_tensorcnt(0);
        }
        __syncthreads();

        for (int st = 0; st < 4; ++st) {        // 16-key subtiles
            const int key0 = st * 16;
            v8f s = (v8f){0,0,0,0,0,0,0,0};
#pragma unroll
            for (int dk = 0; dk < 4; ++dk) {    // contract head_dim 128
                v16bf bk = *(const v16bf*)&Ks[cur][key0 + lh][dk * 32 + hi * 16];
                s = __builtin_amdgcn_wmma_f32_16x16x32_bf16(
                        false, qf[dk], false, bk, (short)0, s, false, false);
            }
            // sigmoid, rowsum accumulation, stage P tile as bf16 in LDS
#pragma unroll
            for (int r = 0; r < 8; ++r) {
                float p = 1.0f / (1.0f + __expf(-s[r] * ATT_SCALE));
                int m = r + hi * 8;
                atomicAdd(&rs[wave][m], p);
                Ps[wave][m][(st & 1) * 16 + lh] = (bf16)p;
            }
            if (st & 1) {                       // 32 keys staged -> P @ V
                v16bf pa = *(const v16bf*)&Ps[wave][lh][hi * 16];
                const int kb = (st - 1) * 16;
#pragma unroll
                for (int nt = 0; nt < 8; ++nt) {
                    v16bf bv;
#pragma unroll
                    for (int i = 0; i < 16; ++i)
                        bv[i] = Vs[cur][kb + hi * 16 + i][nt * 16 + lh];
                    acc[nt] = __builtin_amdgcn_wmma_f32_16x16x32_bf16(
                                  false, pa, false, bv, (short)0, acc[nt],
                                  false, false);
                }
            }
        }
        __syncthreads();                 // done reading before DMA overwrites
    }

    // epilogue: normalize, blend-weight, atomic accumulate into ctx
#pragma unroll
    for (int r = 0; r < 8; ++r) {
        const int m = r + hi * 8;
        const int srow = qs + m;
        const int rr = srow - win_start;
        float wt = (rr < 256)
                       ? ((w == 0) ? 1.0f : (float)rr * (1.0f / 255.0f))
                       : ((float)(511 - rr) * (1.0f / 255.0f));
        float inv = wt / rs[wave][m];
#pragma unroll
        for (int nt = 0; nt < 8; ++nt) {
            int col = nt * 16 + lh;
            atomicAdd(&ctx[((size_t)b * SEQ + srow) * HIDDEN + h * HD + col],
                      acc[nt][r] * inv);
        }
    }
}

// ---------------------------------------------------------------------------
// Host launcher
// ---------------------------------------------------------------------------
extern "C" void kernel_launch(void* const* d_in, const int* in_sizes, int n_in,
                              void* d_out, int out_size, void* d_ws, size_t ws_size,
                              hipStream_t stream)
{
    (void)in_sizes; (void)n_in; (void)out_size; (void)ws_size;
    const float* x   = (const float*)d_in[0];
    const float* wq  = (const float*)d_in[1];
    const float* wk  = (const float*)d_in[2];
    const float* wv  = (const float*)d_in[3];
    const float* wo  = (const float*)d_in[4];
    const float* kcw = (const float*)d_in[5];
    const float* vcw = (const float*)d_in[6];
    const float* kdw = (const float*)d_in[7];
    const float* vdw = (const float*)d_in[8];
    float* out = (float*)d_out;

    // workspace carve-out (256B aligned)
    size_t cur = 0;
    auto carve = [&](size_t bytes) -> char* {
        char* p = (char*)d_ws + cur;
        cur = (cur + bytes + 255) & ~(size_t)255;
        return p;
    };
    const size_t nX  = (size_t)MTOT * HIDDEN;   // 16.7M
    const size_t nKV = (size_t)MTOT * HD;       // 1.05M

    bf16*  xb   = (bf16*)carve(nX * 2);
    bf16*  wqb  = (bf16*)carve((size_t)HIDDEN * HIDDEN * 2);
    bf16*  wkb  = (bf16*)carve((size_t)HD * HIDDEN * 2);
    bf16*  wvb  = (bf16*)carve((size_t)HD * HIDDEN * 2);
    bf16*  wob  = (bf16*)carve((size_t)HIDDEN * HIDDEN * 2);
    float* Qf   = (float*)carve(nX * 4);
    float* Kf   = (float*)carve(nKV * 4);
    float* Vf   = (float*)carve(nKV * 4);
    bf16*  Qb   = (bf16*)carve(nX * 2);
    float* Kr   = (float*)carve(nKV * 4);
    float* Kd   = (float*)carve(nKV * 4);
    float* Vd   = (float*)carve(nKV * 4);
    bf16*  Kq   = (bf16*)carve(nKV * 2);
    bf16*  Vq   = (bf16*)carve(nKV * 2);
    float* ctx  = (float*)carve(nX * 4);
    bf16*  ctxb = (bf16*)carve(nX * 2);
    unsigned* mm = (unsigned*)carve(64);

    const int TB = 256;
    auto gsz = [](size_t n) { return (unsigned)((n + 255) / 256); };

    // 1) convert inputs to bf16
    cvt_f32_bf16_kernel<<<gsz(nX), TB, 0, stream>>>(x, xb, nX);
    cvt_f32_bf16_kernel<<<gsz((size_t)HIDDEN*HIDDEN), TB, 0, stream>>>(wq, wqb, (size_t)HIDDEN*HIDDEN);
    cvt_f32_bf16_kernel<<<gsz((size_t)HD*HIDDEN), TB, 0, stream>>>(wk, wkb, (size_t)HD*HIDDEN);
    cvt_f32_bf16_kernel<<<gsz((size_t)HD*HIDDEN), TB, 0, stream>>>(wv, wvb, (size_t)HD*HIDDEN);
    cvt_f32_bf16_kernel<<<gsz((size_t)HIDDEN*HIDDEN), TB, 0, stream>>>(wo, wob, (size_t)HIDDEN*HIDDEN);
    init_mm_kernel<<<1, 32, 0, stream>>>(mm);
    fill_f32_kernel<<<gsz(nX), TB, 0, stream>>>(ctx, 0.0f, nX);

    // 2) projections: Q = x@wq^T, K = x@wk^T, V = x@wv^T
    gemm_bf16_wt_kernel<<<dim3(HIDDEN/128, MTOT/128), TB, 0, stream>>>(xb, wqb, Qf, MTOT, HIDDEN, HIDDEN);
    gemm_bf16_wt_kernel<<<dim3(HD/128,     MTOT/128), TB, 0, stream>>>(xb, wkb, Kf, MTOT, HD, HIDDEN);
    gemm_bf16_wt_kernel<<<dim3(HD/128,     MTOT/128), TB, 0, stream>>>(xb, wvb, Vf, MTOT, HD, HIDDEN);

    // 3) RoPE (Q -> bf16, K -> f32 for MLA path)
    rope_q_kernel<<<gsz(nX), TB, 0, stream>>>(Qf, Qb, nX);
    rope_k_kernel<<<gsz(nKV), TB, 0, stream>>>(Kf, Kr, nKV);

    // 4) MLA compress/decompress + global min/max
    compress_kernel<<<MTOT, 128, 0, stream>>>(Kr, kcw, kdw, Kd, mm);
    compress_kernel<<<MTOT, 128, 0, stream>>>(Vf, vcw, vdw, Vd, mm + 2);

    // 5) 8-bit quant-dequant -> bf16 K/V
    quant_kernel<<<gsz(nKV), TB, 0, stream>>>(Kd, Kq, mm,     nKV);
    quant_kernel<<<gsz(nKV), TB, 0, stream>>>(Vd, Vq, mm + 2, nKV);

    // 6) windowed attention with cross-fade accumulation into ctx
    attn_kernel<<<BATCH * HEADS * NWIN * 4, TB, 0, stream>>>(Qb, Kq, Vq, ctx);

    // 7) output projection: out = ctx @ wo^T
    cvt_f32_bf16_kernel<<<gsz(nX), TB, 0, stream>>>(ctx, ctxb, nX);
    gemm_bf16_wt_kernel<<<dim3(HIDDEN/128, MTOT/128), TB, 0, stream>>>(ctxb, wob, out, MTOT, HIDDEN, HIDDEN);
}
